// DIENModel_17257178596032
// MI455X (gfx1250) — compile-verified, matrix operand order checked by reading, not run on hardware
//
#include <hip/hip_runtime.h>
#include <hip/hip_bf16.h>
#include <cstddef>
#include <cstdint>

// ---------------------------------------------------------------------------
// CDNA5 (gfx1250) DIEN: wave32, WMMA f32_16x16x32_f16 GEMMs, TDM-staged LDS
// weights for the sequential GRU/AUGRU scans.
// ---------------------------------------------------------------------------

typedef __attribute__((ext_vector_type(16))) _Float16 v16h;
typedef __attribute__((ext_vector_type(8)))  float    v8f;
typedef __attribute__((ext_vector_type(4)))  float    f32x4;
typedef __attribute__((ext_vector_type(4)))  unsigned int u32x4;
typedef __attribute__((ext_vector_type(8)))  int      i32x8;
typedef __attribute__((ext_vector_type(4)))  int      i32x4;

union FragU { v16h v; f32x4 q[2]; _Float16 h[16]; };

static constexpr int B_   = 256;
static constexpr int T_   = 512;
static constexpr int E_   = 128;
static constexpr int H_   = 128;
static constexpr int G3_  = 384;   // 3*H
static constexpr int CATE = 64;    // E/2

__device__ __forceinline__ v8f wmma16(v16h a, v16h b, v8f c) {
  // D = A(16x32 f16) * B(32x16 f16) + C(16x16 f32)
  return __builtin_amdgcn_wmma_f32_16x16x32_f16(false, a, false, b,
                                                (short)0, c, false, false);
}

// A/B fragment load: row-major [rows][ld] f16 source. Per ISA 7.12.2,
// lane r=lane&15 holds row (row0+r); hi=lane>>4 selects K sub-groups:
// elems 0..7 <- K = k0 + hi*8 + [0..7], elems 8..15 <- K = k0+16+hi*8+[0..7].
__device__ __forceinline__ v16h load_frag(const _Float16* base, int ld,
                                          int row0, int k0, int lane) {
  const int r  = lane & 15;
  const int hi = (lane >> 4) & 1;
  const _Float16* p = base + (size_t)(row0 + r) * (size_t)ld + k0 + hi * 8;
  FragU f;
  f.q[0] = *reinterpret_cast<const f32x4*>(p);
  f.q[1] = *reinterpret_cast<const f32x4*>(p + 16);
  return f.v;
}

__device__ __forceinline__ float sigmoidf_(float x) {
  return 1.0f / (1.0f + __expf(-x));
}

// ---------------------------------------------------------------------------
// Tensor Data Mover: one-shot 1-D global->LDS DMA of (n8 * 8) bytes.
// D# per ISA cdna5 ch.8: group0 {count=1, lds_addr, global_addr, type=2},
// group1 {data_size=8B, tensor_dim0=n8, tile_dim0=n8, tile_dim1=1, strides=n8}.
// Issue from one wave; EXEC is ignored for tensor ops. TENSORcnt-tracked.
// ---------------------------------------------------------------------------
#if __has_builtin(__builtin_amdgcn_tensor_load_to_lds)
#define DIEN_HAVE_TDM 1
#else
#define DIEN_HAVE_TDM 0
#endif

#if DIEN_HAVE_TDM
__device__ __forceinline__ void tdm_load_1d(const void* gsrc, void* ldst,
                                            unsigned n8 /* 8-byte units */) {
  const unsigned long long ga = (unsigned long long)(uintptr_t)gsrc;
  const unsigned lds = (unsigned)(uintptr_t)ldst;   // low 32 bits = LDS offset
  u32x4 g0;
  g0.x = 1u;                                          // count=1, user mode
  g0.y = lds;                                         // lds_addr
  g0.z = (unsigned)(ga & 0xFFFFFFFFu);                // global_addr[31:0]
  g0.w = (unsigned)((ga >> 32) & 0x01FFFFFFu) | (2u << 30);  // [56:32] | type=2
  i32x8 g1;
  g1[0] = (int)(3u << 16);                            // data_size=3 (8 bytes)
  g1[1] = (int)((n8 & 0xFFFFu) << 16);                // tensor_dim0[15:0]
  g1[2] = (int)(((n8 >> 16) & 0xFFFFu) | (1u << 16)); // dim0[31:16] | dim1[15:0]=1
  g1[3] = (int)((n8 & 0xFFFFu) << 16);                // dim1[31:16]=0 | tile_dim0
  g1[4] = 1;                                          // tile_dim1=1, tile_dim2=0
  g1[5] = (int)n8;                                    // dim0_stride[31:0]
  g1[6] = (int)((n8 & 0xFFFFu) << 16);                // dim0_str[47:32]=0|dim1_str[15:0]
  g1[7] = (int)(n8 >> 16);                            // dim1_stride[47:16]
  i32x4 g2 = {0, 0, 0, 0};
  i32x4 g3 = {0, 0, 0, 0};
#if defined(__clang_major__) && __clang_major__ >= 23
  i32x8 g4 = {0, 0, 0, 0, 0, 0, 0, 0};
  __builtin_amdgcn_tensor_load_to_lds(g0, g1, g2, g3, g4, 0);
#else
  __builtin_amdgcn_tensor_load_to_lds(g0, g1, g2, g3, 0);
#endif
}
#endif

// Stage `nhalf` f16 elems from global to LDS (TDM if available, else coop copy).
__device__ __forceinline__ void stage_weights(const _Float16* g, _Float16* l,
                                              int nhalf, int tid) {
#if DIEN_HAVE_TDM
  if (tid < 32) {
    tdm_load_1d(g, l, (unsigned)(nhalf / 4));   // 4 halfs = 8 bytes
#if __has_builtin(__builtin_amdgcn_s_wait_tensorcnt)
    __builtin_amdgcn_s_wait_tensorcnt(0);
#endif
  }
#else
  for (int i = tid * 8; i < nhalf; i += 256 * 8)
    *reinterpret_cast<f32x4*>(l + i) = *reinterpret_cast<const f32x4*>(g + i);
#endif
}

// ---------------------------------------------------------------------------
// K0: f32 -> f16 weight conversion
// ---------------------------------------------------------------------------
__global__ void dien_f2h(const float* __restrict__ in, _Float16* __restrict__ out, int n) {
  int i = blockIdx.x * blockDim.x + threadIdx.x;
  if (i < n) out[i] = (_Float16)in[i];
}

// ---------------------------------------------------------------------------
// K1: candidate / category gather (+f16 copy of candidate)
// ---------------------------------------------------------------------------
__global__ void dien_prep(const int* __restrict__ cand_id, const int* __restrict__ cand_cat,
                          const float* __restrict__ item_emb, const float* __restrict__ cat_emb,
                          float* __restrict__ cand_f, _Float16* __restrict__ cand_h,
                          float* __restrict__ cat_f) {
  int b = blockIdx.x, tid = threadIdx.x;
  if (tid < E_) {
    float v = item_emb[(size_t)cand_id[b] * E_ + tid];
    cand_f[(size_t)b * E_ + tid] = v;
    cand_h[(size_t)b * E_ + tid] = (_Float16)v;
  }
  if (tid < CATE) {
    cat_f[(size_t)b * CATE + tid] = cat_emb[(size_t)cand_cat[b] * CATE + tid];
  }
}

// ---------------------------------------------------------------------------
// K2: behavior embedding gather -> f16 [B*T][E]
// ---------------------------------------------------------------------------
__global__ void dien_gather(const int* __restrict__ ids, const float* __restrict__ emb,
                            _Float16* __restrict__ out, int nrows) {
  int g = blockIdx.x * blockDim.x + threadIdx.x;       // one thread per 4 elems
  if (g >= nrows * (E_ / 4)) return;
  int row = g >> 5;
  int c   = (g & 31) * 4;
  f32x4 v = *reinterpret_cast<const f32x4*>(emb + (size_t)ids[row] * E_ + c);
  _Float16* o = out + (size_t)row * E_ + c;
  o[0] = (_Float16)v.x; o[1] = (_Float16)v.y; o[2] = (_Float16)v.z; o[3] = (_Float16)v.w;
}

// ---------------------------------------------------------------------------
// K3: xW = behavior_emb @ Wih^T + bih   (M=B*T, K=128, N=384), WMMA
// ---------------------------------------------------------------------------
__global__ void __launch_bounds__(256)
dien_gemm_xw(const _Float16* __restrict__ X, const _Float16* __restrict__ W,
             const float* __restrict__ bias, float* __restrict__ Y) {
  const int wave = threadIdx.x >> 5, lane = threadIdx.x & 31;
  const int r = lane & 15, hi = (lane >> 4) & 1;
  const int m0 = blockIdx.x * 128 + wave * 16;

  v16h a0 = load_frag(X, E_, m0,  0, lane);
  v16h a1 = load_frag(X, E_, m0, 32, lane);
  v16h a2 = load_frag(X, E_, m0, 64, lane);
  v16h a3 = load_frag(X, E_, m0, 96, lane);

  for (int nt = 0; nt < G3_ / 16; ++nt) {
    const int n0 = nt * 16;
    v8f acc = {};
    acc = wmma16(a0, load_frag(W, E_, n0,  0, lane), acc);
    acc = wmma16(a1, load_frag(W, E_, n0, 32, lane), acc);
    acc = wmma16(a2, load_frag(W, E_, n0, 64, lane), acc);
    acc = wmma16(a3, load_frag(W, E_, n0, 96, lane), acc);
    const int col = n0 + r;
    const float bv = bias[col];
#pragma unroll
    for (int v = 0; v < 8; ++v)
      Y[(size_t)(m0 + v + hi * 8) * G3_ + col] = acc[v] + bv;
  }
}

// ---------------------------------------------------------------------------
// K4: GRU scan. 16 blocks x 256 threads; block owns 16 batch rows.
// Whh (96 KB f16) TDM-staged into LDS once; h kept in LDS (f32 master + f16
// WMMA copy); gh = h @ Whh^T via WMMA each step.
// Dynamic LDS: Whh 98304 | sh_h 4096 | sh_hf 8192 | sh_gh 24576 = 135168 B.
// ---------------------------------------------------------------------------
__global__ void __launch_bounds__(256)
dien_gru(const float* __restrict__ xW, const float* __restrict__ bhh,
         const _Float16* __restrict__ Whh, const unsigned char* __restrict__ mask,
         float* __restrict__ hs_f, _Float16* __restrict__ hs_h) {
  extern __shared__ char smem_gru[];
  _Float16* sh_whh = (_Float16*)smem_gru;                       // 384*128 f16
  _Float16* sh_h   = (_Float16*)(smem_gru + 98304);             // 16*128 f16
  float*    sh_hf  = (float*)   (smem_gru + 98304 + 4096);      // 16*128 f32
  float*    sh_gh  = (float*)   (smem_gru + 98304 + 4096 + 8192); // 16*384 f32

  const int tid = threadIdx.x, wave = tid >> 5, lane = tid & 31;
  const int r = lane & 15, hi = (lane >> 4) & 1;
  const int b0 = blockIdx.x * 16;

  stage_weights(Whh, sh_whh, G3_ * H_, tid);
  for (int i = tid; i < 16 * H_; i += 256) { sh_h[i] = (_Float16)0.0f; sh_hf[i] = 0.0f; }
  __syncthreads();

  for (int t = 0; t < T_; ++t) {
    // gh tile: 24 n-tiles over 8 waves (3 each), B-frags from LDS
#pragma unroll
    for (int j = 0; j < 3; ++j) {
      const int n0 = (wave * 3 + j) * 16;
      v8f acc = {};
#pragma unroll
      for (int kt = 0; kt < 4; ++kt) {
        v16h a = load_frag((const _Float16*)sh_h, H_, 0, kt * 32, lane);
        v16h b = load_frag((const _Float16*)sh_whh, H_, n0, kt * 32, lane);
        acc = wmma16(a, b, acc);
      }
      const int col = n0 + r;
#pragma unroll
      for (int v = 0; v < 8; ++v) sh_gh[(v + hi * 8) * G3_ + col] = acc[v];
    }
    __syncthreads();

    for (int i = tid; i < 16 * H_; i += 256) {
      const int bl = i >> 7, c = i & 127;
      const size_t row = (size_t)(b0 + bl) * T_ + t;
      const float hr = sh_gh[bl * G3_ + c]         + bhh[c];
      const float hz = sh_gh[bl * G3_ + H_ + c]    + bhh[H_ + c];
      const float hn = sh_gh[bl * G3_ + 2*H_ + c]  + bhh[2*H_ + c];
      const float xr = xW[row * G3_ + c];
      const float xz = xW[row * G3_ + H_ + c];
      const float xn = xW[row * G3_ + 2*H_ + c];
      // prefetch next timestep's xW row segments (global_prefetch_b8)
      if (((c & 31) == 0) && (t + 1 < T_)) {
        __builtin_prefetch(&xW[row * G3_ + G3_ + c], 0, 0);
        __builtin_prefetch(&xW[row * G3_ + G3_ + H_ + c], 0, 0);
        __builtin_prefetch(&xW[row * G3_ + G3_ + 2*H_ + c], 0, 0);
      }
      const float rg = sigmoidf_(xr + hr);
      const float zg = sigmoidf_(xz + hz);
      const float ng = tanhf(xn + rg * hn);
      const float hprev = sh_hf[bl * H_ + c];
      const float hnew  = (1.0f - zg) * ng + zg * hprev;
      const float m  = mask[row] ? 1.0f : 0.0f;
      const float ho = hnew * m;
      hs_f[row * H_ + c] = ho;
      hs_h[row * H_ + c] = (_Float16)ho;
      sh_hf[bl * H_ + c] = hnew;
      sh_h [bl * H_ + c] = (_Float16)hnew;
    }
    __syncthreads();
  }
}

// ---------------------------------------------------------------------------
// K5: attention scores. att_in = [hs, cand, hs*cand] built on the fly per
// 8-half fragment group; s = prelu(att_in @ W1^T + b1); score = s.w2 + b2.
// ---------------------------------------------------------------------------
__device__ __forceinline__ f32x4 attin8(const _Float16* hrow, const _Float16* crow, int k) {
  if (k < 128) return *reinterpret_cast<const f32x4*>(hrow + k);
  if (k < 256) return *reinterpret_cast<const f32x4*>(crow + (k - 128));
  const int kk = k - 256;
  union { f32x4 q; _Float16 h[8]; } a, c, o;
  a.q = *reinterpret_cast<const f32x4*>(hrow + kk);
  c.q = *reinterpret_cast<const f32x4*>(crow + kk);
#pragma unroll
  for (int i = 0; i < 8; ++i) o.h[i] = a.h[i] * c.h[i];
  return o.q;
}

__global__ void __launch_bounds__(256)
dien_attn(const _Float16* __restrict__ hs_h, const _Float16* __restrict__ cand_h,
          const _Float16* __restrict__ W1, const float* __restrict__ b1,
          const float* __restrict__ alpha_p, const float* __restrict__ w2,
          const float* __restrict__ b2, const unsigned char* __restrict__ mask,
          float* __restrict__ scores) {
  __shared__ float s_s[128 * 64];
  const int tid = threadIdx.x, wave = tid >> 5, lane = tid & 31;
  const int r = lane & 15, hi = (lane >> 4) & 1;
  const int m0 = blockIdx.x * 128;
  const int mw = wave * 16;
  const float alpha = *alpha_p;

  const int myrow = m0 + mw + r;                 // per-lane A row
  const _Float16* hrow = hs_h  + (size_t)myrow * H_;
  const _Float16* crow = cand_h + (size_t)(myrow / T_) * E_;

#pragma unroll
  for (int nt = 0; nt < 4; ++nt) {
    v8f acc = {};
#pragma unroll
    for (int kt = 0; kt < 12; ++kt) {
      FragU af;
      const int kA = kt * 32 + hi * 8;
      af.q[0] = attin8(hrow, crow, kA);
      af.q[1] = attin8(hrow, crow, kA + 16);
      v16h bf = load_frag(W1, G3_, nt * 16, kt * 32, lane);
      acc = wmma16(af.v, bf, acc);
    }
    const int col = nt * 16 + r;
    const float bb = b1[col], wv = w2[col];
#pragma unroll
    for (int v = 0; v < 8; ++v) {
      float x = acc[v] + bb;
      x = (x >= 0.0f) ? x : alpha * x;
      s_s[(mw + v + hi * 8) * 64 + col] = x * wv;
    }
  }
  __syncthreads();
  if (tid < 128) {
    float sum = 0.0f;
#pragma unroll 8
    for (int c = 0; c < 64; ++c) sum += s_s[tid * 64 + c];
    const size_t row = (size_t)m0 + tid;
    const float val = sum + b2[0];
    scores[row] = mask[row] ? val : -__builtin_inff();
  }
}

// ---------------------------------------------------------------------------
// K6: softmax over T per batch row
// ---------------------------------------------------------------------------
__global__ void __launch_bounds__(256)
dien_softmax(const float* __restrict__ scores, float* __restrict__ att) {
  __shared__ float red[256];
  const int b = blockIdx.x, tid = threadIdx.x;
  const float* s = scores + (size_t)b * T_;
  float m = -__builtin_inff();
  for (int t = tid; t < T_; t += 256) m = fmaxf(m, s[t]);
  red[tid] = m; __syncthreads();
  for (int st = 128; st > 0; st >>= 1) { if (tid < st) red[tid] = fmaxf(red[tid], red[tid + st]); __syncthreads(); }
  m = red[0]; __syncthreads();
  float sum = 0.0f;
  for (int t = tid; t < T_; t += 256) sum += __expf(s[t] - m);
  red[tid] = sum; __syncthreads();
  for (int st = 128; st > 0; st >>= 1) { if (tid < st) red[tid] += red[tid + st]; __syncthreads(); }
  const float inv = 1.0f / red[0];
  for (int t = tid; t < T_; t += 256) att[(size_t)b * T_ + t] = __expf(s[t] - m) * inv;
}

// ---------------------------------------------------------------------------
// K7: auxiliary head: relu(hs[:, :-1] @ aux_w1^T + b1) @ aux_w2 + b2
// ---------------------------------------------------------------------------
__global__ void __launch_bounds__(256)
dien_aux(const _Float16* __restrict__ hs_h, const _Float16* __restrict__ W1,
         const float* __restrict__ b1, const float* __restrict__ w2,
         const float* __restrict__ b2, float* __restrict__ out) {
  __shared__ float s_p[128 * 16];
  const int tid = threadIdx.x, wave = tid >> 5, lane = tid & 31;
  const int r = lane & 15, hi = (lane >> 4) & 1;
  const int m0 = blockIdx.x * 128;
  const int mw = wave * 16;

  v16h a0 = load_frag(hs_h, H_, m0 + mw,  0, lane);
  v16h a1 = load_frag(hs_h, H_, m0 + mw, 32, lane);
  v16h a2 = load_frag(hs_h, H_, m0 + mw, 64, lane);
  v16h a3 = load_frag(hs_h, H_, m0 + mw, 96, lane);

  float part[8] = {};
#pragma unroll
  for (int nt = 0; nt < 8; ++nt) {
    const int n0 = nt * 16;
    v8f acc = {};
    acc = wmma16(a0, load_frag(W1, H_, n0,  0, lane), acc);
    acc = wmma16(a1, load_frag(W1, H_, n0, 32, lane), acc);
    acc = wmma16(a2, load_frag(W1, H_, n0, 64, lane), acc);
    acc = wmma16(a3, load_frag(W1, H_, n0, 96, lane), acc);
    const int col = n0 + r;
    const float bb = b1[col], wv = w2[col];
#pragma unroll
    for (int v = 0; v < 8; ++v) {
      float x = acc[v] + bb;
      x = (x > 0.0f) ? x : 0.0f;
      part[v] += x * wv;
    }
  }
#pragma unroll
  for (int v = 0; v < 8; ++v) s_p[(mw + v + hi * 8) * 16 + r] = part[v];
  __syncthreads();
  if (tid < 128) {
    float sum = b2[0];
#pragma unroll
    for (int i = 0; i < 16; ++i) sum += s_p[tid * 16 + i];
    const size_t row = (size_t)m0 + tid;
    const int t = (int)(row % T_), bb = (int)(row / T_);
    if (t < T_ - 1) out[(size_t)bb * (T_ - 1) + t] = sum;
  }
}

// ---------------------------------------------------------------------------
// K8: AUGRU scan. 16 blocks x 256 threads; 16 batch rows per block.
// Wz|Wr|Wh (192 KB f16, contiguous) TDM-staged into LDS once.
// Dynamic LDS: weights 196608 | comb 8192 | z/r/ht/hf 4*8192 = 237568 B.
// ---------------------------------------------------------------------------
__global__ void __launch_bounds__(256)
dien_augru(const _Float16* __restrict__ hs_h, const float* __restrict__ att,
           const _Float16* __restrict__ Wzrw,   // Wz | Wr | Wh, each 128x256 f16
           const float* __restrict__ bz, const float* __restrict__ br,
           const float* __restrict__ bh, float* __restrict__ evolved) {
  extern __shared__ char smem_au[];
  _Float16* sh_w    = (_Float16*)smem_au;                   // 3 * 128*256 f16
  _Float16* s_comb  = (_Float16*)(smem_au + 196608);        // 16*256 f16
  float*    s_z     = (float*)(smem_au + 196608 + 8192);
  float*    s_r     = (float*)(smem_au + 196608 + 8192 + 8192);
  float*    s_ht    = (float*)(smem_au + 196608 + 8192 + 2 * 8192);
  float*    s_hf    = (float*)(smem_au + 196608 + 8192 + 3 * 8192);
  const _Float16* sh_wz = sh_w;
  const _Float16* sh_wr = sh_w + 32768;
  const _Float16* sh_wh = sh_w + 65536;

  const int tid = threadIdx.x, wave = tid >> 5, lane = tid & 31;
  const int r = lane & 15, hi = (lane >> 4) & 1;
  const int b0 = blockIdx.x * 16;

  stage_weights(Wzrw, sh_w, 3 * H_ * 256, tid);
  for (int i = tid; i < 16 * H_; i += 256) {
    s_hf[i] = 0.0f;
    s_comb[(i >> 7) * 256 + 128 + (i & 127)] = (_Float16)0.0f;
  }
  __syncthreads();

  for (int t = 0; t < T_; ++t) {
    // x part of combined
    for (int i = tid; i < 16 * H_; i += 256) {
      const int bl = i >> 7, c = i & 127;
      s_comb[bl * 256 + c] = hs_h[((size_t)(b0 + bl) * T_ + t) * H_ + c];
    }
    __syncthreads();

    // phase 1: Z and R (16 n-tiles over 8 waves)
#pragma unroll
    for (int j = 0; j < 2; ++j) {
      const int ntile = wave * 2 + j;                   // 0..15
      const _Float16* W  = (ntile < 8) ? sh_wz : sh_wr;
      const float*    bi = (ntile < 8) ? bz : br;
      float*          ds = (ntile < 8) ? s_z : s_r;
      const int n0 = (ntile & 7) * 16;
      v8f acc = {};
#pragma unroll
      for (int kt = 0; kt < 8; ++kt) {
        v16h a = load_frag((const _Float16*)s_comb, 256, 0, kt * 32, lane);
        v16h b = load_frag(W, 256, n0, kt * 32, lane);
        acc = wmma16(a, b, acc);
      }
      const int col = n0 + r;
      const float bb = bi[col];
#pragma unroll
      for (int v = 0; v < 8; ++v)
        ds[(v + hi * 8) * H_ + col] = sigmoidf_(acc[v] + bb);
    }
    __syncthreads();

    // combined_r: upper half <- r * h
    for (int i = tid; i < 16 * H_; i += 256) {
      const int bl = i >> 7, c = i & 127;
      s_comb[bl * 256 + 128 + c] = (_Float16)(s_r[bl * H_ + c] * s_hf[bl * H_ + c]);
    }
    __syncthreads();

    // phase 2: h_tilde (8 n-tiles, 1 per wave)
    {
      const int n0 = wave * 16;
      v8f acc = {};
#pragma unroll
      for (int kt = 0; kt < 8; ++kt) {
        v16h a = load_frag((const _Float16*)s_comb, 256, 0, kt * 32, lane);
        v16h b = load_frag(sh_wh, 256, n0, kt * 32, lane);
        acc = wmma16(a, b, acc);
      }
      const int col = n0 + r;
      const float bb = bh[col];
#pragma unroll
      for (int v = 0; v < 8; ++v)
        s_ht[(v + hi * 8) * H_ + col] = tanhf(acc[v] + bb);
    }
    __syncthreads();

    // h update (attention-gated) + refresh combined h-part for next step
    for (int i = tid; i < 16 * H_; i += 256) {
      const int bl = i >> 7, c = i & 127;
      const float a  = att[(size_t)(b0 + bl) * T_ + t];
      const float zp = a * s_z[bl * H_ + c];
      const float hn = (1.0f - zp) * s_hf[bl * H_ + c] + zp * s_ht[bl * H_ + c];
      s_hf[bl * H_ + c] = hn;
      s_comb[bl * 256 + 128 + c] = (_Float16)hn;
    }
    __syncthreads();
  }

  for (int i = tid; i < 16 * H_; i += 256) {
    const int bl = i >> 7, c = i & 127;
    evolved[(size_t)(b0 + bl) * H_ + c] = s_hf[bl * H_ + c];
  }
}

// ---------------------------------------------------------------------------
// K9: concat + 3-layer MLP (LayerNorm+PReLU) + sigmoid head. One block per b.
// ---------------------------------------------------------------------------
__device__ void mlp_layer(const float* w, const float* bias, const float* g,
                          const float* be, float alpha, int K, int N,
                          float* xin, float* xout, float* red, int tid) {
  for (int o = tid; o < N; o += 256) {
    float s = bias[o];
    const float* wr = w + (size_t)o * K;
    for (int k = 0; k < K; ++k) s += xin[k] * wr[k];
    xout[o] = s;
  }
  __syncthreads();
  float part = 0.0f;
  for (int o = tid; o < N; o += 256) part += xout[o];
  red[tid] = part; __syncthreads();
  for (int st = 128; st > 0; st >>= 1) { if (tid < st) red[tid] += red[tid + st]; __syncthreads(); }
  const float mu = red[0] / N; __syncthreads();
  part = 0.0f;
  for (int o = tid; o < N; o += 256) { float d = xout[o] - mu; part += d * d; }
  red[tid] = part; __syncthreads();
  for (int st = 128; st > 0; st >>= 1) { if (tid < st) red[tid] += red[tid + st]; __syncthreads(); }
  const float inv = rsqrtf(red[0] / N + 1e-5f); __syncthreads();
  for (int o = tid; o < N; o += 256) {
    float v = (xout[o] - mu) * inv * g[o] + be[o];
    xin[o] = (v >= 0.0f) ? v : alpha * v;
  }
  __syncthreads();
}

__global__ void __launch_bounds__(256)
dien_mlp(const float* __restrict__ evolved, const float* __restrict__ cand,
         const float* __restrict__ cat, const float* __restrict__ dense,
         const float* __restrict__ dpw, const float* __restrict__ dpb,
         const float* w0, const float* b0, const float* g0, const float* be0, const float* a0,
         const float* w1, const float* b1, const float* g1, const float* be1, const float* a1,
         const float* w2, const float* b2, const float* g2, const float* be2, const float* a2,
         const float* __restrict__ hw, const float* __restrict__ hb,
         float* __restrict__ preds) {
  __shared__ float xin[512];
  __shared__ float xout[512];
  __shared__ float red[256];
  const int b = blockIdx.x, tid = threadIdx.x;

  for (int i = tid; i < 352; i += 256) {
    float v;
    if (i < 128)       v = evolved[(size_t)b * H_ + i];
    else if (i < 256)  v = cand[(size_t)b * E_ + (i - 128)];
    else if (i < 320)  v = cat[(size_t)b * CATE + (i - 256)];
    else {
      const int j = i - 320;
      float s = dpb[j];
      for (int k = 0; k < 5; ++k) s += dense[(size_t)b * 5 + k] * dpw[j * 5 + k];
      v = s;
    }
    xin[i] = v;
  }
  __syncthreads();

  mlp_layer(w0, b0, g0, be0, *a0, 352, 512, xin, xout, red, tid);
  mlp_layer(w1, b1, g1, be1, *a1, 512, 256, xin, xout, red, tid);
  mlp_layer(w2, b2, g2, be2, *a2, 256, 128, xin, xout, red, tid);

  if (tid < 3) {
    float s = hb[tid];
    for (int k = 0; k < 128; ++k) s += xin[k] * hw[tid * 128 + k];
    preds[(size_t)b * 3 + tid] = sigmoidf_(s);
  }
}

// ---------------------------------------------------------------------------
// Launcher
// ---------------------------------------------------------------------------
extern "C" void kernel_launch(void* const* d_in, const int* in_sizes, int n_in,
                              void* d_out, int out_size, void* d_ws, size_t ws_size,
                              hipStream_t stream) {
  (void)in_sizes; (void)n_in; (void)out_size; (void)ws_size;
  const int*   behavior_ids = (const int*)  d_in[0];
  const int*   candidate_id = (const int*)  d_in[1];
  const int*   candidate_ct = (const int*)  d_in[2];
  const float* dense        = (const float*)d_in[3];
  const unsigned char* mask = (const unsigned char*)d_in[4];
  const float* item_emb     = (const float*)d_in[5];
  const float* cat_emb      = (const float*)d_in[6];
  const float* gru_Wih      = (const float*)d_in[7];
  const float* gru_Whh      = (const float*)d_in[8];
  const float* gru_bih      = (const float*)d_in[9];
  const float* gru_bhh      = (const float*)d_in[10];
  const float* aux_w1       = (const float*)d_in[11];
  const float* aux_b1       = (const float*)d_in[12];
  const float* aux_w2       = (const float*)d_in[13];
  const float* aux_b2       = (const float*)d_in[14];
  const float* att_w1       = (const float*)d_in[15];
  const float* att_b1       = (const float*)d_in[16];
  const float* att_alpha    = (const float*)d_in[17];
  const float* att_w2       = (const float*)d_in[18];
  const float* att_b2       = (const float*)d_in[19];
  const float* Wz           = (const float*)d_in[20];
  const float* bz           = (const float*)d_in[21];
  const float* Wr           = (const float*)d_in[22];
  const float* br           = (const float*)d_in[23];
  const float* Wh           = (const float*)d_in[24];
  const float* bh           = (const float*)d_in[25];
  const float* dp_w         = (const float*)d_in[26];
  const float* dp_b         = (const float*)d_in[27];
  const float* m0w = (const float*)d_in[28]; const float* m0b  = (const float*)d_in[29];
  const float* m0g = (const float*)d_in[30]; const float* m0be = (const float*)d_in[31];
  const float* m0a = (const float*)d_in[32];
  const float* m1w = (const float*)d_in[33]; const float* m1b  = (const float*)d_in[34];
  const float* m1g = (const float*)d_in[35]; const float* m1be = (const float*)d_in[36];
  const float* m1a = (const float*)d_in[37];
  const float* m2w = (const float*)d_in[38]; const float* m2b  = (const float*)d_in[39];
  const float* m2g = (const float*)d_in[40]; const float* m2be = (const float*)d_in[41];
  const float* m2a = (const float*)d_in[42];
  const float* head_w = (const float*)d_in[43];
  const float* head_b = (const float*)d_in[44];

  float* out_preds = (float*)d_out;            // 256*3
  float* out_aux   = (float*)d_out + B_ * 3;   // 256*511

  // workspace layout
  char* ws = (char*)d_ws;
  size_t off = 0;
  auto alloc = [&](size_t bytes) -> char* {
    off = (off + 255) & ~(size_t)255;
    char* p = ws + off;
    off += bytes;
    return p;
  };
  const size_t MROWS = (size_t)B_ * T_;      // 131072
  _Float16* wih_h  = (_Float16*)alloc((size_t)G3_ * E_ * 2);
  _Float16* whh_h  = (_Float16*)alloc((size_t)G3_ * H_ * 2);
  _Float16* auxw1h = (_Float16*)alloc((size_t)E_ * H_ * 2);
  _Float16* attw1h = (_Float16*)alloc((size_t)64 * G3_ * 2);
  _Float16* zrw_h  = (_Float16*)alloc((size_t)3 * H_ * 256 * 2);  // Wz|Wr|Wh contiguous
  float*    cand_f = (float*)   alloc((size_t)B_ * E_ * 4);
  _Float16* cand_h = (_Float16*)alloc((size_t)B_ * E_ * 2);
  float*    cat_f  = (float*)   alloc((size_t)B_ * CATE * 4);
  float*    evolved= (float*)   alloc((size_t)B_ * H_ * 4);
  _Float16* beh_h  = (_Float16*)alloc(MROWS * E_ * 2);
  float*    xw     = (float*)   alloc(MROWS * G3_ * 4);
  float*    hs_f   = (float*)   alloc(MROWS * H_ * 4);
  _Float16* hs_h   = (_Float16*)alloc(MROWS * H_ * 2);
  float*    scores = (float*)   alloc(MROWS * 4);
  float*    attp   = (float*)   alloc(MROWS * 4);

  auto f2h = [&](const float* src, _Float16* dst, int n) {
    dien_f2h<<<(n + 255) / 256, 256, 0, stream>>>(src, dst, n);
  };
  f2h(gru_Wih, wih_h,  G3_ * E_);
  f2h(gru_Whh, whh_h,  G3_ * H_);
  f2h(aux_w1,  auxw1h, E_ * H_);
  f2h(att_w1,  attw1h, 64 * G3_);
  f2h(Wz,      zrw_h,              H_ * 256);
  f2h(Wr,      zrw_h + 32768,      H_ * 256);
  f2h(Wh,      zrw_h + 65536,      H_ * 256);

  dien_prep<<<B_, 128, 0, stream>>>(candidate_id, candidate_ct, item_emb, cat_emb,
                                    cand_f, cand_h, cat_f);

  dien_gather<<<(int)((MROWS * (E_ / 4) + 255) / 256), 256, 0, stream>>>(
      behavior_ids, item_emb, beh_h, (int)MROWS);

  dien_gemm_xw<<<(int)(MROWS / 128), 256, 0, stream>>>(beh_h, wih_h, gru_bih, xw);

  // dynamic LDS: Whh(98304) + h16(4096) + hf(8192) + gh(24576)
  dien_gru<<<B_ / 16, 256, 135168, stream>>>(xw, gru_bhh, whh_h, mask, hs_f, hs_h);

  dien_attn<<<(int)(MROWS / 128), 256, 0, stream>>>(hs_h, cand_h, attw1h, att_b1,
                                                    att_alpha, att_w2, att_b2, mask, scores);

  dien_softmax<<<B_, 256, 0, stream>>>(scores, attp);

  dien_aux<<<(int)(MROWS / 128), 256, 0, stream>>>(hs_h, auxw1h, aux_b1, aux_w2,
                                                   aux_b2, out_aux);

  // dynamic LDS: weights(196608) + comb(8192) + 4*8192
  dien_augru<<<B_ / 16, 256, 237568, stream>>>(hs_h, attp, zrw_h, bz, br, bh, evolved);

  dien_mlp<<<B_, 256, 0, stream>>>(evolved, cand_f, cat_f, dense, dp_w, dp_b,
                                   m0w, m0b, m0g, m0be, m0a,
                                   m1w, m1b, m1g, m1be, m1a,
                                   m2w, m2b, m2g, m2be, m2a,
                                   head_w, head_b, out_preds);
}
